// AntiSymmetricConv_28492813041843
// MI455X (gfx1250) — matrix operand order repaired and checked by reading, not compiled, not run on hardware
//
#include <hip/hip_runtime.h>
#include <math.h>

#define DIM   256
#define GAMMA 0.1f
#define EPS   0.1f

typedef __attribute__((ext_vector_type(16))) _Float16 v16h;
typedef __attribute__((ext_vector_type(8)))  _Float16 v8h;
typedef __attribute__((ext_vector_type(8)))  float    v8f;

// ---------------- precompute kernels ----------------

__global__ void deg_init_kernel(float* __restrict__ deg, int n) {
    int i = blockIdx.x * blockDim.x + threadIdx.x;
    if (i < n) deg[i] = 1.0f;   // self loop contributes 1
}

__global__ void deg_accum_kernel(const int* __restrict__ dst, float* __restrict__ deg, int e) {
    int i = blockIdx.x * blockDim.x + threadIdx.x;
    if (i < e) atomicAdd(&deg[dst[i]], 1.0f);
}

__global__ void deg_to_dinv_kernel(float* __restrict__ deg, int n) {
    int i = blockIdx.x * blockDim.x + threadIdx.x;
    if (i < n) {
        float d = deg[i];
        deg[i] = (d > 0.0f) ? rsqrtf(d) : 0.0f;
    }
}

// Wt_phi[n*D + k] = (f16) W_phi[k*D + n]   (B stored column-major so K is contiguous)
__global__ void make_wtphi_kernel(const float* __restrict__ Wphi, _Float16* __restrict__ Wt) {
    int i = blockIdx.x * blockDim.x + threadIdx.x;
    if (i < DIM * DIM) {
        int n = i / DIM, k = i % DIM;
        Wt[i] = (_Float16)Wphi[k * DIM + n];
    }
}

// x @ (W - W^T - g I)^T  ==  x @ M with M[k,j] = W[j,k] - W[k,j] - g*d(j,k)
// Bt[j*D + k] = M[k,j] = W[j,k] - W[k,j] - g*d(j,k)
__global__ void make_bt_kernel(const float* __restrict__ W, _Float16* __restrict__ Bt) {
    int i = blockIdx.x * blockDim.x + threadIdx.x;
    if (i < DIM * DIM) {
        int j = i / DIM, k = i % DIM;
        float v = W[j * DIM + k] - W[k * DIM + j] - ((j == k) ? GAMMA : 0.0f);
        Bt[i] = (_Float16)v;
    }
}

// ---------------- WMMA fragment helpers ----------------

// A fragment (16x32 f16): lane holds row m = lane&15, contiguous K = ko..ko+15,
// ko = (lane>>4)*16.  arow pre-offset by row*D + ko; convert fp32 -> f16 on the fly.
__device__ __forceinline__ v16h load_a_frag(const float* __restrict__ arow, int kb) {
    float4 a0 = *(const float4*)(arow + kb);
    float4 a1 = *(const float4*)(arow + kb + 4);
    float4 a2 = *(const float4*)(arow + kb + 8);
    float4 a3 = *(const float4*)(arow + kb + 12);
    v16h a;
    a[0]  = (_Float16)a0.x; a[1]  = (_Float16)a0.y; a[2]  = (_Float16)a0.z; a[3]  = (_Float16)a0.w;
    a[4]  = (_Float16)a1.x; a[5]  = (_Float16)a1.y; a[6]  = (_Float16)a1.z; a[7]  = (_Float16)a1.w;
    a[8]  = (_Float16)a2.x; a[9]  = (_Float16)a2.y; a[10] = (_Float16)a2.z; a[11] = (_Float16)a2.w;
    a[12] = (_Float16)a3.x; a[13] = (_Float16)a3.y; a[14] = (_Float16)a3.z; a[15] = (_Float16)a3.w;
    return a;
}

// B fragment (32x16 f16): lane holds col n = lane&15, contiguous K = (lane>>4)*16 .. +15.
// brow pre-offset by n*D + (lane>>4)*16 in the transposed (col-major) weight copy.
__device__ __forceinline__ v16h load_b_frag(const _Float16* __restrict__ brow, int kb) {
    v8h b0 = *(const v8h*)(brow + kb);
    v8h b1 = *(const v8h*)(brow + kb + 8);
    v16h b;
#pragma unroll
    for (int i = 0; i < 8; ++i) { b[i] = b0[i]; b[8 + i] = b1[i]; }
    return b;
}

// ---------------- shared GEMM core: one wave -> 16 rows x 64 cols (4 tiles) ----------------
// Block = 128 threads (4 waves) covers 16 rows x 256 cols. A fragment loaded once per
// K-step and reused by 4 independent WMMAs (hides WMMA WAR hazard slots, 4x less A VMEM).

struct GemmAcc { v8f acc[4]; };

__device__ __forceinline__ GemmAcc gemm_core(const float* __restrict__ A,
                                             const _Float16* __restrict__ Bt,
                                             int row0, int col0, int lane) {
    const int mn = lane & 15;
    const int ko = (lane >> 4) * 16;

    const float* arow = A + (size_t)(row0 + mn) * DIM + ko;
    const _Float16* brow0 = Bt + (size_t)(col0 +  0 + mn) * DIM + (lane >> 4) * 16;
    const _Float16* brow1 = Bt + (size_t)(col0 + 16 + mn) * DIM + (lane >> 4) * 16;
    const _Float16* brow2 = Bt + (size_t)(col0 + 32 + mn) * DIM + (lane >> 4) * 16;
    const _Float16* brow3 = Bt + (size_t)(col0 + 48 + mn) * DIM + (lane >> 4) * 16;

    GemmAcc g;
    g.acc[0] = (v8f){}; g.acc[1] = (v8f){}; g.acc[2] = (v8f){}; g.acc[3] = (v8f){};

#pragma unroll
    for (int kb = 0; kb < DIM; kb += 32) {
        v16h a  = load_a_frag(arow, kb);
        v16h b0 = load_b_frag(brow0, kb);
        v16h b1 = load_b_frag(brow1, kb);
        v16h b2 = load_b_frag(brow2, kb);
        v16h b3 = load_b_frag(brow3, kb);
        g.acc[0] = __builtin_amdgcn_wmma_f32_16x16x32_f16(false, a, false, b0, (short)0, g.acc[0], false, false);
        g.acc[1] = __builtin_amdgcn_wmma_f32_16x16x32_f16(false, a, false, b1, (short)0, g.acc[1], false, false);
        g.acc[2] = __builtin_amdgcn_wmma_f32_16x16x32_f16(false, a, false, b2, (short)0, g.acc[2], false, false);
        g.acc[3] = __builtin_amdgcn_wmma_f32_16x16x32_f16(false, a, false, b3, (short)0, g.acc[3], false, false);
    }
    return g;
}

// ---------------- GEMM 1: h0 = x @ W_phi, fold self-loop norms ----------------
// Writes h0s[n] = h0[n]*dinv[n]  and  agg[n] = h0[n]*dinv[n]^2 (self-loop message).

__global__ void __launch_bounds__(128)
gemm1_wmma_kernel(const float* __restrict__ A, const _Float16* __restrict__ Bt,
                  const float* __restrict__ dinv,
                  float* __restrict__ h0s, float* __restrict__ agg) {
    const int lane = threadIdx.x & 31;
    const int wave = threadIdx.x >> 5;
    const int row0 = blockIdx.x * 16;
    const int col0 = wave * 64;

    GemmAcc g = gemm_core(A, Bt, row0, col0, lane);

    const int mbase = (lane >> 4) * 8;
#pragma unroll
    for (int t = 0; t < 4; ++t) {
        int col = col0 + t * 16 + (lane & 15);
#pragma unroll
        for (int r = 0; r < 8; ++r) {
            int row   = row0 + mbase + r;
            float di  = dinv[row];
            float hv  = g.acc[t][r] * di;
            size_t o  = (size_t)row * DIM + col;
            h0s[o] = hv;
            agg[o] = hv * di;
        }
    }
}

// ---------------- edge scatter: agg[dst] += h0s[src] * dinv[dst] ----------------
// one wave per edge; lane covers channels lane*4..+3 and +128..+131 (two float4 loads)

__global__ void __launch_bounds__(256)
scatter_edges_kernel(const int* __restrict__ src, const int* __restrict__ dst,
                     const float* __restrict__ dinv, const float* __restrict__ h0s,
                     float* __restrict__ agg, int e) {
    int edge = blockIdx.x * 8 + (threadIdx.x >> 5);
    if (edge >= e) return;
    int lane = threadIdx.x & 31;
    int s = src[edge];
    int d = dst[edge];
    float w = dinv[d];
    const float* hs = h0s + (size_t)s * DIM;
    float*       ad = agg + (size_t)d * DIM;
    int c = lane * 4;
    float4 v0 = *(const float4*)(hs + c);
    float4 v1 = *(const float4*)(hs + c + 128);
    atomicAdd(ad + c + 0,   v0.x * w);
    atomicAdd(ad + c + 1,   v0.y * w);
    atomicAdd(ad + c + 2,   v0.z * w);
    atomicAdd(ad + c + 3,   v0.w * w);
    atomicAdd(ad + c + 128, v1.x * w);
    atomicAdd(ad + c + 129, v1.y * w);
    atomicAdd(ad + c + 130, v1.z * w);
    atomicAdd(ad + c + 131, v1.w * w);
}

// ---------------- GEMM 2 + epilogue: out = x + eps*tanh(x@M + agg + bias) ----------------

__global__ void __launch_bounds__(128)
gemm2_wmma_kernel(const float* __restrict__ A, const _Float16* __restrict__ Bt,
                  const float* __restrict__ agg, const float* __restrict__ bias,
                  float* __restrict__ out) {
    const int lane = threadIdx.x & 31;
    const int wave = threadIdx.x >> 5;
    const int row0 = blockIdx.x * 16;
    const int col0 = wave * 64;

    GemmAcc g = gemm_core(A, Bt, row0, col0, lane);

    const int mbase = (lane >> 4) * 8;
#pragma unroll
    for (int t = 0; t < 4; ++t) {
        int col  = col0 + t * 16 + (lane & 15);
        float bb = bias[col];
#pragma unroll
        for (int r = 0; r < 8; ++r) {
            int row  = row0 + mbase + r;
            size_t o = (size_t)row * DIM + col;
            float z  = g.acc[t][r] + agg[o] + bb;
            out[o]   = A[o] + EPS * tanhf(z);
        }
    }
}

// ---------------- host launcher ----------------

extern "C" void kernel_launch(void* const* d_in, const int* in_sizes, int n_in,
                              void* d_out, int out_size, void* d_ws, size_t ws_size,
                              hipStream_t stream) {
    const float* x0   = (const float*)d_in[0];   // [N, 256] fp32
    const int*   ei   = (const int*)  d_in[1];   // [2, E] int32 (src row, dst row)
    const float* W    = (const float*)d_in[2];   // [256, 256]
    const float* Wphi = (const float*)d_in[3];   // [256, 256]
    const float* bias = (const float*)d_in[4];   // [256]

    const int N = in_sizes[0] / DIM;
    const int E = in_sizes[1] / 2;
    const int* src = ei;
    const int* dst = ei + E;

    char*  ws  = (char*)d_ws;
    size_t off = 0;
    auto carve = [&](size_t bytes) -> char* {
        char* p = ws + off;
        off += (bytes + 255) & ~(size_t)255;
        return p;
    };
    float*    h0s   = (float*)   carve((size_t)N * DIM * sizeof(float));
    float*    agg   = (float*)   carve((size_t)N * DIM * sizeof(float));
    float*    xA    = (float*)   carve((size_t)N * DIM * sizeof(float));
    float*    xB    = (float*)   carve((size_t)N * DIM * sizeof(float));
    float*    dinv  = (float*)   carve((size_t)N * sizeof(float));
    _Float16* WtPhi = (_Float16*)carve((size_t)DIM * DIM * sizeof(_Float16));
    _Float16* Bt    = (_Float16*)carve((size_t)DIM * DIM * sizeof(_Float16));

    // ---- one-time precompute (re-derived every call; deterministic) ----
    deg_init_kernel   <<<(N + 255) / 256, 256, 0, stream>>>(dinv, N);
    deg_accum_kernel  <<<(E + 255) / 256, 256, 0, stream>>>(dst, dinv, E);
    deg_to_dinv_kernel<<<(N + 255) / 256, 256, 0, stream>>>(dinv, N);
    make_wtphi_kernel <<<(DIM * DIM + 255) / 256, 256, 0, stream>>>(Wphi, WtPhi);
    make_bt_kernel    <<<(DIM * DIM + 255) / 256, 256, 0, stream>>>(W, Bt);

    // ---- 4 iterations, ping-pong x; last write goes to d_out ----
    dim3 ggrid(N / 16);        // 3125 row tiles; 4 waves * 64 cols = full 256 cols per block
    dim3 gblk(128);
    dim3 sgrid((E + 7) / 8);   // one wave (32 lanes) per edge, 8 edges per block
    dim3 sblk(256);

    const float* xs = x0;
    float* xd[4] = { xA, xB, xA, (float*)d_out };
    for (int it = 0; it < 4; ++it) {
        gemm1_wmma_kernel   <<<ggrid, gblk, 0, stream>>>(xs, WtPhi, dinv, h0s, agg);
        scatter_edges_kernel<<<sgrid, sblk, 0, stream>>>(src, dst, dinv, h0s, agg, E);
        gemm2_wmma_kernel   <<<ggrid, gblk, 0, stream>>>(xs, Bt, agg, bias, xd[it]);
        xs = xd[it];
    }
}
